// TopKPool_515396076383
// MI455X (gfx1250) — matrix operand order repaired and checked by reading, not compiled, not run on hardware
//
#include <hip/hip_runtime.h>
#include <math.h>

// ---------------------------------------------------------------------------
// TopKPool for MI455X (gfx1250, wave32).
//  out = concat( X_pooled (4096x256 f32), A_pooled (4096x4096 f32) )
//  A_pooled computed as A[idx,:] @ A[:,idx] via v_wmma_f32_16x16x32_f16
//  (f16 inputs in [0,1), f32 accumulation over K=8192).
// ---------------------------------------------------------------------------

typedef __attribute__((ext_vector_type(16))) _Float16 v16h;
typedef __attribute__((ext_vector_type(8)))  _Float16 v8h;
typedef __attribute__((ext_vector_type(4)))  _Float16 v4h;
typedef __attribute__((ext_vector_type(8)))  float    v8f;
typedef __attribute__((ext_vector_type(4)))  float    f4;

#define NN     8192
#define FF     256
#define KSEL   4096
#define KSTEP  32
#define NKT    (NN / KSTEP)          // 256 k-tiles
#define LDSTR  48                    // halfs per LDS row (96B: 16B-aligned, bank-spread)

// ---------------------------------------------------------------------------
// Kernel 1: y = X @ (p / ||p||).  One wave per row, 8 rows per 256-thr block.
// ---------------------------------------------------------------------------
__global__ __launch_bounds__(256) void k_score(const float* __restrict__ X,
                                               const float* __restrict__ p,
                                               float* __restrict__ y) {
  __shared__ float sp[FF];
  __shared__ float red[256];
  __shared__ float sinv;
  const int t = threadIdx.x;
  float pv = p[t];
  sp[t] = pv;
  red[t] = pv * pv;
  __syncthreads();
  for (int s = 128; s > 0; s >>= 1) {
    if (t < s) red[t] += red[t + s];
    __syncthreads();
  }
  if (t == 0) sinv = 1.0f / sqrtf(red[0]);
  __syncthreads();
  const float inv = sinv;

  const int wave = t >> 5, lane = t & 31;
  const int row = blockIdx.x * 8 + wave;
  const float* xr = X + (size_t)row * FF;
  float acc = 0.0f;
#pragma unroll
  for (int f = 0; f < FF; f += 32) acc += xr[f + lane] * sp[f + lane];
#pragma unroll
  for (int off = 16; off > 0; off >>= 1) acc += __shfl_down(acc, off, 32);
  if (lane == 0) y[row] = acc * inv;
}

// ---------------------------------------------------------------------------
// Kernel 2: exact top-k (k=4096) by rank, jax tie-break (value desc, idx asc),
// output indices ascending. Single block, all-LDS O(N^2) rank select + scan.
// ---------------------------------------------------------------------------
__global__ __launch_bounds__(1024) void k_select(const float* __restrict__ y,
                                                 int* __restrict__ idx) {
  __shared__ float sy[NN];
  __shared__ int   scnt[1024];
  const int t = threadIdx.x;
  for (int i = t; i < NN; i += 1024) sy[i] = y[i];
  __syncthreads();

  const int base = t * 8;
  float v[8];
  int   rank[8];
#pragma unroll
  for (int c = 0; c < 8; c++) { v[c] = sy[base + c]; rank[c] = 0; }

  for (int j = 0; j < NN; j++) {
    const float w = sy[j];
#pragma unroll
    for (int c = 0; c < 8; c++)
      rank[c] += (w > v[c]) || ((w == v[c]) && (j < base + c));
  }

  int flag[8], cnt = 0;
#pragma unroll
  for (int c = 0; c < 8; c++) { flag[c] = (rank[c] < KSEL); cnt += flag[c]; }

  scnt[t] = cnt;
  __syncthreads();
  // Hillis-Steele inclusive scan over 1024 counts
  for (int off = 1; off < 1024; off <<= 1) {
    int add = (t >= off) ? scnt[t - off] : 0;
    __syncthreads();
    scnt[t] += add;
    __syncthreads();
  }
  int pos = scnt[t] - cnt;  // exclusive
#pragma unroll
  for (int c = 0; c < 8; c++)
    if (flag[c]) idx[pos++] = base + c;
}

// ---------------------------------------------------------------------------
// Kernel 3: X_pooled[i,:] = X[idx[i],:] * tanh(y[idx[i]])
// ---------------------------------------------------------------------------
__global__ __launch_bounds__(256) void k_xpool(const float* __restrict__ X,
                                               const float* __restrict__ y,
                                               const int* __restrict__ idx,
                                               float* __restrict__ out) {
  const int i = blockIdx.x;           // 0..4095
  const int t = threadIdx.x;          // 0..255
  const int g = idx[i];
  const float gate = tanhf(y[g]);
  out[(size_t)i * FF + t] = X[(size_t)g * FF + t] * gate;
}

// ---------------------------------------------------------------------------
// Kernel 4: A_pooled = A[idx,:] @ A[:,idx]   (4096x8192)@(8192x4096), f32 acc.
// Workgroup tile 256x128 (8 waves of 64x64), v_wmma_f32_16x16x32_f16.
// blockIdx.x = row block so concurrently-launched workgroups share one
// gathered column block (L2 reuse of low-utilization gather lines).
// ---------------------------------------------------------------------------
__global__ __launch_bounds__(256) void k_gemm(const float* __restrict__ A,
                                              const int* __restrict__ idx,
                                              float* __restrict__ C) {
  __shared__ _Float16 As[256 * LDSTR];   // [m][k] k-contiguous, 24 KB
  __shared__ _Float16 Bs[128 * LDSTR];   // [n][k] k-contiguous, 12 KB

  const int t    = threadIdx.x;
  const int wave = t >> 5;
  const int lane = t & 31;
  const int row0 = blockIdx.x * 256;     // x-major launch: concurrent wgs
  const int col0 = blockIdx.y * 128;     //   share the same column block
  const int wr   = (wave >> 1) * 64;     // wave row offset in tile (0..192)
  const int wc   = (wave & 1)  * 64;     // wave col offset in tile (0,64)

  v8f acc[4][4];
#pragma unroll
  for (int mi = 0; mi < 4; mi++)
#pragma unroll
    for (int ni = 0; ni < 4; ni++) acc[mi][ni] = (v8f)(0.0f);

  // --- staging assignments -------------------------------------------------
  // A tile: thread handles rows {p*32 + t/8 : p=0..7}, k-chunk (t%8)*4 (float4)
  const int ar = t >> 3;                 // 0..31
  const int ak = (t & 7) * 4;            // 0,4,..,28
  int grow[8];
#pragma unroll
  for (int p = 0; p < 8; p++) grow[p] = idx[row0 + p * 32 + ar];
  // B tile: thread handles col t&127, 16 k-values starting at (t>>7)*16
  const int bj   = t & 127;
  const int bk0  = (t >> 7) * 16;
  const int gcol = idx[col0 + bj];

  f4    a_st[8];
  float b_st[16];

  // --- prologue: global-load k-tile 0 into registers -----------------------
#pragma unroll
  for (int p = 0; p < 8; p++)
    a_st[p] = *(const f4*)(A + (size_t)grow[p] * NN + ak);
#pragma unroll
  for (int q = 0; q < 16; q++)
    b_st[q] = A[(size_t)(bk0 + q) * NN + gcol];

  for (int kt = 0; kt < NKT; kt++) {
    __syncthreads();                     // LDS free (prev tile fully consumed)

    // registers -> LDS (f32 -> f16 convert)
#pragma unroll
    for (int p = 0; p < 8; p++) {
      v4h hv;
#pragma unroll
      for (int e = 0; e < 4; e++) hv[e] = (_Float16)a_st[p][e];
      *(v4h*)(&As[(p * 32 + ar) * LDSTR + ak]) = hv;
    }
    {
      v8h lo, hi;
#pragma unroll
      for (int e = 0; e < 8; e++) { lo[e] = (_Float16)b_st[e]; hi[e] = (_Float16)b_st[8 + e]; }
      _Float16* pb = &Bs[bj * LDSTR + bk0];
      *(v8h*)pb = lo;
      *((v8h*)pb + 1) = hi;
    }
    __syncthreads();                     // tile kt visible to all waves

    // issue global loads for tile kt+1 (latency hidden under the WMMAs)
    if (kt + 1 < NKT) {
      const int k0 = (kt + 1) * KSTEP;
#pragma unroll
      for (int p = 0; p < 8; p++)
        a_st[p] = *(const f4*)(A + (size_t)grow[p] * NN + k0 + ak);
#pragma unroll
      for (int q = 0; q < 16; q++)
        b_st[q] = A[(size_t)(k0 + bk0 + q) * NN + gcol];
    }

    // --- fragment loads (ISA VGPR layouts) + 16 WMMAs per wave -------------
    v16h bf[4];
#pragma unroll
    for (int ni = 0; ni < 4; ni++) {
      // B 32x16: lane<16 holds K{0..7,16..23} of col N=lane, lane>=16 K{8..15,24..31}
      const int n  = wc + ni * 16 + (lane & 15);
      const int kb = (lane >> 4) * 8;
      const _Float16* pB = &Bs[n * LDSTR + kb];
      v16h r;
      ((f4*)&r)[0] = *(const f4*)pB;         // 8 halfs
      ((f4*)&r)[1] = *(const f4*)(pB + 16);  // K kb+16..kb+23
      bf[ni] = r;
    }
#pragma unroll
    for (int mi = 0; mi < 4; mi++) {
      // A 16x32: lane<16 holds K{0..7,16..23}, lane>=16 K{8..15,24..31}
      const int m  = wr + mi * 16 + (lane & 15);
      const int kb = (lane >> 4) * 8;
      const _Float16* pA = &As[m * LDSTR + kb];
      v16h af;
      ((f4*)&af)[0] = *(const f4*)pA;        // K kb..kb+7
      ((f4*)&af)[1] = *(const f4*)(pA + 16); // K kb+16..kb+23
#pragma unroll
      for (int ni = 0; ni < 4; ni++)
        acc[mi][ni] = __builtin_amdgcn_wmma_f32_16x16x32_f16(
            false, af, false, bf[ni], (short)0, acc[mi][ni], false, false);
    }
  }

  // --- epilogue: C/D layout -> global (VGPR r: M=r lanes0-15, M=8+r lanes16-31)
#pragma unroll
  for (int mi = 0; mi < 4; mi++) {
#pragma unroll
    for (int ni = 0; ni < 4; ni++) {
      const v8f c  = acc[mi][ni];
      const int col   = col0 + wc + ni * 16 + (lane & 15);
      const int rbase = row0 + wr + mi * 16 + (lane >> 4) * 8;
#pragma unroll
      for (int e = 0; e < 8; e++)
        C[(size_t)(rbase + e) * KSEL + col] = c[e];
    }
  }
}

// ---------------------------------------------------------------------------
extern "C" void kernel_launch(void* const* d_in, const int* in_sizes, int n_in,
                              void* d_out, int out_size, void* d_ws, size_t ws_size,
                              hipStream_t stream) {
  (void)in_sizes; (void)n_in; (void)out_size; (void)ws_size;
  const float* X = (const float*)d_in[0];   // (8192, 256)
  const float* A = (const float*)d_in[1];   // (8192, 8192)
  const float* p = (const float*)d_in[2];   // (256, 1)
  float* out = (float*)d_out;               // X_pooled | A_pooled

  float* y   = (float*)d_ws;                               // 8192 f32
  int*   idx = (int*)((char*)d_ws + NN * sizeof(float));   // 4096 i32

  k_score <<<NN / 8, 256, 0, stream>>>(X, p, y);
  k_select<<<1, 1024, 0, stream>>>(y, idx);
  k_xpool <<<KSEL, 256, 0, stream>>>(X, y, idx, out);

  dim3 grid(KSEL / 256, KSEL / 128);       // 16 row-blocks x 32 col-blocks
  k_gemm<<<grid, 256, 0, stream>>>(A, idx, out + (size_t)KSEL * FF);
}
